// TextRNN_8426725834773
// MI455X (gfx1250) — matrix operand order repaired and checked by reading, not compiled, not run on hardware
//
#include <hip/hip_runtime.h>

// ---------------------------------------------------------------------------
// Types for CDNA5 WMMA (wave32): v_wmma_f32_16x16x32_bf16
// ---------------------------------------------------------------------------
typedef __attribute__((ext_vector_type(8)))  float        v8f;
typedef __attribute__((ext_vector_type(16))) __bf16       v16bf;
typedef __attribute__((ext_vector_type(8)))  __bf16       v8bf;
typedef __attribute__((ext_vector_type(4)))  unsigned int u32x4;
typedef __attribute__((ext_vector_type(8)))  int          i32x8;
typedef __attribute__((ext_vector_type(4)))  int          i32x4;

// Problem constants (match reference)
#define Vv 10000
#define Ee 128
#define Hh 256
#define Bb 32
#define Ss 512
#define Mrows (Ss * Bb)        // 16384 rows, M index = t*32 + b
#define Ngate 1024             // 4*H

// Tensor Data Mover availability (this toolchain: 6-arg builtin)
#if defined(__HIP_DEVICE_COMPILE__) && __has_builtin(__builtin_amdgcn_tensor_load_to_lds)
#define HAS_TDM 1
#else
#define HAS_TDM 0
#endif

__device__ __forceinline__ __bf16 f2bf(float f) {
    union { float f; unsigned u; } v; v.f = f;
    unsigned r = (v.u + 0x7FFFu + ((v.u >> 16) & 1u)) >> 16;
    unsigned short s = (unsigned short)r;
    return __builtin_bit_cast(__bf16, s);
}
__device__ __forceinline__ float sigmoidf_(float x) { return 1.f / (1.f + __expf(-x)); }

// A fragment (16x32 bf16, row-major source): lanes 0-15 -> rows 0-15, K {0..7,16..23};
// lanes 16-31 -> rows 0-15, K {8..15,24..31}. Two contiguous 16B loads per lane.
__device__ __forceinline__ v16bf load_a_frag(const __bf16* A, int lda, int m0, int k0, int lane) {
    const __bf16* p = A + (size_t)(m0 + (lane & 15)) * lda + k0 + ((lane >> 4) << 3);
    v8bf lo = *(const v8bf*)(p);
    v8bf hi = *(const v8bf*)(p + 16);
    return __builtin_shufflevector(lo, hi, 0,1,2,3,4,5,6,7,8,9,10,11,12,13,14,15);
}

// B fragment from packed weights. Pack layout: P[((k>>5)*N + n)*32 + (k&31)]
// Two 16B reads per lane: K {8h..8h+7, 16+8h..16+8h+7}.
__device__ __forceinline__ v16bf load_b_frag(const __bf16* Bp, int N, int kt, int n0, int lane) {
    const __bf16* p = Bp + (((size_t)kt * N + n0 + (lane & 15)) << 5) + ((lane >> 4) << 3);
    v8bf lo = *(const v8bf*)(p);
    v8bf hi = *(const v8bf*)(p + 16);
    return __builtin_shufflevector(lo, hi, 0,1,2,3,4,5,6,7,8,9,10,11,12,13,14,15);
}

__device__ __forceinline__ v8f wmma_bf16(v16bf a, v16bf b, v8f c) {
    return __builtin_amdgcn_wmma_f32_16x16x32_bf16(false, a, false, b, (short)0, c, false, false);
}

#if HAS_TDM
// TDM: DMA one [32 x 1024] f32 row-block of Xpre into LDS (2-D tile descriptor).
__device__ __forceinline__ void tdm_load_x(const float* gsrc, void* lds_dst) {
    unsigned long long ga = (unsigned long long)(const void*)gsrc;
    unsigned lds_off = (unsigned)(unsigned long long)lds_dst;  // low 32 bits = LDS byte offset
    u32x4 g0;
    g0[0] = 1u;                                            // count=1 (valid descriptor)
    g0[1] = lds_off;                                       // lds_addr
    g0[2] = (unsigned)(ga & 0xFFFFFFFFu);                  // global_addr[31:0]
    g0[3] = ((unsigned)(ga >> 32) & 0x01FFFFFFu) | (2u << 30); // global_addr[56:32] | type=2
    i32x8 g1;
    g1[0] = (2 << 16);                 // workgroup_mask=0, data_size=2 (4B)
    g1[1] = (Ngate & 0xFFFF) << 16;    // tensor_dim0[15:0]=1024
    g1[2] = (Mrows & 0xFFFF) << 16;    // tensor_dim0[31:16]=0 | tensor_dim1[15:0]=16384
    g1[3] = (Ngate << 16);             // tensor_dim1[31:16]=0 | tile_dim0=1024
    g1[4] = Bb;                        // tile_dim1=32, tile_dim2=0
    g1[5] = Ngate;                     // tensor_dim0_stride[31:0]=1024
    g1[6] = 0;                         // stride0 hi | stride1 lo
    g1[7] = 0;
    i32x4 g2; g2[0] = 0; g2[1] = 0; g2[2] = 0; g2[3] = 0;
    i32x4 g3; g3[0] = 0; g3[1] = 0; g3[2] = 0; g3[3] = 0;
    i32x8 g4; g4[0] = 0; g4[1] = 0; g4[2] = 0; g4[3] = 0;
    g4[4] = 0; g4[5] = 0; g4[6] = 0; g4[7] = 0;
    __builtin_amdgcn_tensor_load_to_lds(g0, g1, g2, g3, g4, 0);
}
#endif

// ---------------------------------------------------------------------------
// Pack f32 row-major [K][N] weights into bf16 WMMA-B layout.
// ---------------------------------------------------------------------------
__global__ void pack_b_kernel(const float* __restrict__ W, __bf16* __restrict__ P, int K, int N) {
    int idx = blockIdx.x * blockDim.x + threadIdx.x;
    if (idx >= K * N) return;
    int k = idx / N, n = idx - k * N;
    P[(((size_t)(k >> 5) * N + n) << 5) + (k & 31)] = f2bf(W[idx]);
}

// Embedding gather -> bf16 activations xs[M][E], M = t*32 + b
__global__ void embed_kernel(const int* __restrict__ x, const float* __restrict__ emb,
                             __bf16* __restrict__ xs) {
    int idx = blockIdx.x * blockDim.x + threadIdx.x;
    if (idx >= Mrows * Ee) return;
    int m = idx >> 7, e = idx & (Ee - 1);
    int t = m >> 5, b = m & 31;
    int tok = x[b * Ss + t];
    xs[idx] = f2bf(emb[(size_t)tok * Ee + e]);
}

__global__ void zero_kernel(float* p, int n) {
    int i = blockIdx.x * blockDim.x + threadIdx.x;
    if (i < n) p[i] = 0.f;
}

// ---------------------------------------------------------------------------
// Parallel input-half GEMM: C[M][1024] = A[M][K]bf16 @ Bpack + bias  (f32 out)
// ---------------------------------------------------------------------------
__global__ __launch_bounds__(256) void gemm_bias_kernel(
    const __bf16* __restrict__ A, int lda, int Kt,
    const __bf16* __restrict__ Bp, const float* __restrict__ bias,
    float* __restrict__ C) {
    int lane = threadIdx.x & 31, w = threadIdx.x >> 5;
    int m0 = blockIdx.x * 16;
    int n0 = (blockIdx.y * 8 + w) * 16;
    v8f acc;
#pragma unroll
    for (int i = 0; i < 8; ++i) acc[i] = 0.f;
    for (int kt = 0; kt < Kt; ++kt) {
        v16bf a = load_a_frag(A, lda, m0, kt * 32, lane);
        v16bf b = load_b_frag(Bp, Ngate, kt, n0, lane);
        acc = wmma_bf16(a, b, acc);
    }
    int col = n0 + (lane & 15);
    float bv = bias[col];
    int rb = m0 + ((lane >> 4) << 3);
#pragma unroll
    for (int i = 0; i < 8; ++i) C[(size_t)(rb + i) * Ngate + col] = acc[i] + bv;
}

// ---------------------------------------------------------------------------
// Persistent single-workgroup LSTM recurrence. 1024 threads = 32 waves.
// Wave w (m = w>>4, wc = w&15) computes the FOUR matching gate tiles
// (n = g*16 + wc) so each lane owns 8 (row, j) cells: all gate pre-acts plus
// c/h state stay in registers. LDS: h bf16 16KB + double-buffered 2x128KB
// Xpre tile streamed by the Tensor Data Mover (s_wait_tensorcnt sync).
// ---------------------------------------------------------------------------
__global__ __launch_bounds__(1024) void lstm_kernel(
    const float* __restrict__ Xpre, const __bf16* __restrict__ Whp,
    const int* __restrict__ lengths, __bf16* __restrict__ hout) {
    extern __shared__ char smem[];
#if HAS_TDM
    float*  Xb[2];
    Xb[0] = (float*)smem;                         // 32*1024 f32 = 128KB
    Xb[1] = (float*)(smem + 131072);              // 128KB
    __bf16* hb = (__bf16*)(smem + 262144);        // [32][256] bf16 = 16KB
#else
    __bf16* hb = (__bf16*)smem;
#endif

    int tid = threadIdx.x, lane = tid & 31, w = tid >> 5;
    for (int i = tid; i < Bb * Hh; i += 1024)
        hb[i] = __builtin_bit_cast(__bf16, (unsigned short)0);

    const int m0  = (w >> 4) << 4;                   // row block 0 or 16
    const int col = ((w & 15) << 4) + (lane & 15);   // hidden index j
    const int rb  = m0 + ((lane >> 4) << 3);         // this lane's first row

    int   mylen[8];
    float c_reg[8], h_reg[8];
#pragma unroll
    for (int i = 0; i < 8; ++i) {
        mylen[i] = lengths[rb + i];
        c_reg[i] = 0.f;
        h_reg[i] = 0.f;
    }
#if HAS_TDM
    if (w == 0) tdm_load_x(Xpre, Xb[0]);          // prefetch t=0
#endif
    __syncthreads();

    for (int t = 0; t < Ss; ++t) {
#if HAS_TDM
        if (w == 0) {
            if (t + 1 < Ss) {
                tdm_load_x(Xpre + (size_t)(t + 1) * Bb * Ngate, Xb[(t + 1) & 1]);
                __builtin_amdgcn_s_wait_tensorcnt(1);   // tile t landed
            } else {
                __builtin_amdgcn_s_wait_tensorcnt(0);
            }
        }
        __syncthreads();                                // X[t] visible, hb writes of t-1 done
        const float* Xt = Xb[t & 1];
#else
        __syncthreads();
        const float* Xt = Xpre + (size_t)t * Bb * Ngate;
#endif
        // ---- phase A: z = X[t] + h @ Wh, all four gates in registers ----
        v8f zi, zg, zf, zo;
#pragma unroll
        for (int i = 0; i < 8; ++i) {
            const float* xr = Xt + (size_t)(rb + i) * Ngate + col;
            zi[i] = xr[0];
            zg[i] = xr[Hh];
            zf[i] = xr[2 * Hh];
            zo[i] = xr[3 * Hh];
        }
        v16bf afr[8];
#pragma unroll
        for (int kt = 0; kt < 8; ++kt) afr[kt] = load_a_frag(hb, Hh, m0, kt * 32, lane);
        int nb = (w & 15) << 4;
#pragma unroll
        for (int kt = 0; kt < 8; ++kt) zi = wmma_bf16(afr[kt], load_b_frag(Whp, Ngate, kt, nb, lane), zi);
#pragma unroll
        for (int kt = 0; kt < 8; ++kt) zg = wmma_bf16(afr[kt], load_b_frag(Whp, Ngate, kt, Hh + nb, lane), zg);
#pragma unroll
        for (int kt = 0; kt < 8; ++kt) zf = wmma_bf16(afr[kt], load_b_frag(Whp, Ngate, kt, 2 * Hh + nb, lane), zf);
#pragma unroll
        for (int kt = 0; kt < 8; ++kt) zo = wmma_bf16(afr[kt], load_b_frag(Whp, Ngate, kt, 3 * Hh + nb, lane), zo);
        __syncthreads();   // all hb reads complete before phase-B writes

        // ---- phase B: gates (TF order i,g,f,o; forget_bias=1), reg-resident ----
#pragma unroll
        for (int i = 0; i < 8; ++i) {
            int row = rb + i;
            bool act = t < mylen[i];
            float cn = sigmoidf_(zf[i] + 1.f) * c_reg[i] + sigmoidf_(zi[i]) * tanhf(zg[i]);
            float hn = sigmoidf_(zo[i]) * tanhf(cn);
            c_reg[i] = act ? cn : c_reg[i];
            h_reg[i] = act ? hn : h_reg[i];
            hb[row * Hh + col] = f2bf(h_reg[i]);
            hout[((size_t)(t * Bb + row)) * Hh + col] = f2bf(act ? hn : 0.f);
        }
    }
}

// ---------------------------------------------------------------------------
// Fused projection + log-softmax cross-entropy (online logsumexp).
// ---------------------------------------------------------------------------
__global__ __launch_bounds__(256) void loss_kernel(
    const __bf16* __restrict__ H, const __bf16* __restrict__ OWp,
    const float* __restrict__ ob, const int* __restrict__ y,
    float* __restrict__ per_ex) {
    __shared__ float smx[8][16];
    __shared__ float ssm[8][16];
    __shared__ float stgt[16];

    int lane = threadIdx.x & 31, w = threadIdx.x >> 5;
    int m0 = blockIdx.x * 16;
    int rbase = (lane >> 4) << 3;

    if (threadIdx.x < 16) stgt[threadIdx.x] = 0.f;
    __syncthreads();

    int yrow[8];
#pragma unroll
    for (int i = 0; i < 8; ++i) {
        int gr = m0 + rbase + i;
        yrow[i] = y[(gr & 31) * Ss + (gr >> 5)];
    }
    v16bf afr[8];
#pragma unroll
    for (int kt = 0; kt < 8; ++kt) afr[kt] = load_a_frag(H, Hh, m0, kt * 32, lane);

    float mx[8], sm[8];
#pragma unroll
    for (int i = 0; i < 8; ++i) { mx[i] = -__builtin_inff(); sm[i] = 0.f; }

    for (int nt = w; nt < Vv / 16; nt += 8) {   // 625 column tiles
        int n0 = nt * 16;
        v8f acc;
#pragma unroll
        for (int i = 0; i < 8; ++i) acc[i] = 0.f;
#pragma unroll
        for (int kt = 0; kt < 8; ++kt)
            acc = wmma_bf16(afr[kt], load_b_frag(OWp, Vv, kt, n0, lane), acc);
        int col = n0 + (lane & 15);
        float bv = ob[col];
#pragma unroll
        for (int i = 0; i < 8; ++i) {
            float v = acc[i] + bv;
            if (col == yrow[i]) stgt[rbase + i] = v;   // exactly one writer per row
            float nm = fmaxf(mx[i], v);
            sm[i] = sm[i] * __expf(mx[i] - nm) + __expf(v - nm);
            mx[i] = nm;
        }
    }
    for (int off = 1; off < 16; off <<= 1) {
#pragma unroll
        for (int i = 0; i < 8; ++i) {
            float om = __shfl_xor(mx[i], off);
            float os = __shfl_xor(sm[i], off);
            float nm = fmaxf(mx[i], om);
            sm[i] = sm[i] * __expf(mx[i] - nm) + os * __expf(om - nm);
            mx[i] = nm;
        }
    }
    if ((lane & 15) == 0) {
#pragma unroll
        for (int i = 0; i < 8; ++i) { smx[w][rbase + i] = mx[i]; ssm[w][rbase + i] = sm[i]; }
    }
    __syncthreads();
    if (threadIdx.x < 16) {
        int r = threadIdx.x;
        float M = -__builtin_inff(), S = 0.f;
        for (int ww = 0; ww < 8; ++ww) {
            float om = smx[ww][r], os = ssm[ww][r];
            float nm = fmaxf(M, om);
            S = S * __expf(M - nm) + os * __expf(om - nm);
            M = nm;
        }
        float lse = M + __logf(S);
        int gr = m0 + r, b = gr & 31, s = gr >> 5;
        if (y[b * Ss + s] != 0) atomicAdd(&per_ex[b], lse - stgt[r]);
    }
}

__global__ void finalize_kernel(const float* __restrict__ per_ex,
                                const int* __restrict__ len, float* __restrict__ out) {
    int b = threadIdx.x;
    float v = per_ex[b] / (float)len[b];
    for (int off = 16; off; off >>= 1) v += __shfl_down(v, off);
    if (b == 0) out[0] = v / (float)Bb;
}

// ---------------------------------------------------------------------------
extern "C" void kernel_launch(void* const* d_in, const int* in_sizes, int n_in,
                              void* d_out, int out_size, void* d_ws, size_t ws_size,
                              hipStream_t stream) {
    (void)in_sizes; (void)n_in; (void)out_size; (void)ws_size;
    const int*   input_x = (const int*)d_in[0];
    const int*   input_y = (const int*)d_in[1];
    const int*   lengths = (const int*)d_in[2];
    const float* embW    = (const float*)d_in[3];
    const float* k0      = (const float*)d_in[4];
    const float* b0      = (const float*)d_in[5];
    const float* k1      = (const float*)d_in[6];
    const float* b1      = (const float*)d_in[7];
    const float* outW    = (const float*)d_in[8];
    const float* outb    = (const float*)d_in[9];
    float* out = (float*)d_out;

    char* ws = (char*)d_ws;
    size_t off = 0;
    auto alloc = [&](size_t bytes) {
        void* p = ws + off;
        off = (off + bytes + 255) & ~(size_t)255;
        return p;
    };
    __bf16* k0p  = (__bf16*)alloc((size_t)12 * Ngate * 32 * 2);
    __bf16* k1p  = (__bf16*)alloc((size_t)16 * Ngate * 32 * 2);
    __bf16* owp  = (__bf16*)alloc((size_t)8 * Vv * 32 * 2);
    __bf16* xsb  = (__bf16*)alloc((size_t)Mrows * Ee * 2);
    __bf16* h1b  = (__bf16*)alloc((size_t)Mrows * Hh * 2);
    __bf16* h2b  = (__bf16*)alloc((size_t)Mrows * Hh * 2);
    float*  Xbuf = (float*)alloc((size_t)Mrows * Ngate * 4);
    float*  perex = (float*)alloc(Bb * 4);

    // 2x128KB X double buffer + 16KB h buffer (fits CDNA5's 320KB WGP LDS)
    const size_t lstm_smem = 2 * 131072 + (size_t)Bb * Hh * 2;

    pack_b_kernel<<<(384 * Ngate + 255) / 256, 256, 0, stream>>>(k0, k0p, 384, Ngate);
    pack_b_kernel<<<(512 * Ngate + 255) / 256, 256, 0, stream>>>(k1, k1p, 512, Ngate);
    pack_b_kernel<<<(Hh * Vv + 255) / 256, 256, 0, stream>>>(outW, owp, Hh, Vv);
    embed_kernel<<<(Mrows * Ee + 255) / 256, 256, 0, stream>>>(input_x, embW, xsb);
    zero_kernel<<<1, 32, 0, stream>>>(perex, Bb);

    // Layer 0: input half (K=128 -> Kt=4), recurrence (Wh = k0 rows 128..383)
    gemm_bias_kernel<<<dim3(Mrows / 16, 8), 256, 0, stream>>>(xsb, Ee, 4, k0p, b0, Xbuf);
    lstm_kernel<<<1, 1024, lstm_smem, stream>>>(Xbuf, k0p + (size_t)4 * Ngate * 32, lengths, h1b);

    // Layer 1: input half (K=256 -> Kt=8), recurrence (Wh = k1 rows 256..511)
    gemm_bias_kernel<<<dim3(Mrows / 16, 8), 256, 0, stream>>>(h1b, Hh, 8, k1p, b1, Xbuf);
    lstm_kernel<<<1, 1024, lstm_smem, stream>>>(Xbuf, k1p + (size_t)8 * Ngate * 32, lengths, h2b);

    loss_kernel<<<Mrows / 16, 256, 0, stream>>>(h2b, owp, outb, input_y, perex);
    finalize_kernel<<<1, 32, 0, stream>>>(perex, lengths, out);
}
